// EdgeConv_7593502179720
// MI455X (gfx1250) — matrix operand order repaired and checked by reading, not compiled, not run on hardware
//
#include <hip/hip_runtime.h>
#include <hip/hip_bf16.h>

typedef __attribute__((ext_vector_type(16))) __bf16 v16bf;
typedef __attribute__((ext_vector_type(8)))  float  v8f;

#define KNN_K 16
#define C_IN  64
#define D_HID 64

__device__ __forceinline__ unsigned short f2bf(float f) {
  union { float f; unsigned u; } c; c.f = f;
  unsigned r = c.u + 0x7FFFu + ((c.u >> 16) & 1u);   // round-to-nearest-even
  return (unsigned short)(r >> 16);
}

union V16U { uint4 q[2]; v16bf v; };

// A operand: 16xK tile, row-major bf16 source in LDS.
// 16-bit A layout: lanes 0-15 row M=lane, K0..7 in v0-3, K16..23 in v4-7;
// lanes 16-31 same rows, K8..15 / K24..31.  (chunk0 = +16B*laneHi, chunk1 = +32B)
__device__ __forceinline__ v16bf ldsA(const unsigned short* base, int row,
                                      int strideEl, int kt, int laneHi) {
  const unsigned short* p = base + row * strideEl + kt * 32 + laneHi * 8;
  V16U u;
  u.q[0] = *(const uint4*)(p);
  u.q[1] = *(const uint4*)(p + 16);
  return u.v;
}

// B operand: KxN tile fed column-major => read W^T (row-major [N][K]) from LDS.
// B layout: lanes 0-15 col N=lane, K0..7 in v0-3, K16..23 in v4-7; lanes 16-31 K8..15 / K24..31.
__device__ __forceinline__ v16bf ldsB(const unsigned short* base, int col,
                                      int strideEl, int kt, int laneHi) {
  const unsigned short* p = base + col * strideEl + kt * 32 + laneHi * 8;
  V16U u;
  u.q[0] = *(const uint4*)(p);
  u.q[1] = *(const uint4*)(p + 16);
  return u.v;
}

#define WMMA_BF16(a, b, c) \
  __builtin_amdgcn_wmma_f32_16x16x32_bf16(false, (a), false, (b), (short)0, (c), false, false)

// ---------------------------------------------------------------------------
// Kernel 1: brute-force KNN (K=16) over 3-D positions, candidates tiled
// through LDS, per-thread top-16 kept in registers (unsorted + worst slot).
// ---------------------------------------------------------------------------
#define KTILE 2048
#define FLTMAX 3.402823466e38f

__global__ void __launch_bounds__(128) knn_kernel(const float* __restrict__ pos,
                                                  int* __restrict__ idxOut,
                                                  int N) {
  __shared__ float sx[KTILE], sy[KTILE], sz[KTILE];
  const int q = blockIdx.x * 128 + threadIdx.x;
  const float qx = pos[q * 3 + 0];
  const float qy = pos[q * 3 + 1];
  const float qz = pos[q * 3 + 2];

  float bd[KNN_K];
  int   bi[KNN_K];
#pragma unroll
  for (int s = 0; s < KNN_K; ++s) { bd[s] = FLTMAX; bi[s] = 0; }
  float worst = FLTMAX;
  int worstSlot = 0;

  for (int base = 0; base < N; base += KTILE) {
    __syncthreads();
    for (int t = threadIdx.x; t < KTILE; t += 128) {
      const float* p = pos + (size_t)(base + t) * 3;
      sx[t] = p[0]; sy[t] = p[1]; sz[t] = p[2];
    }
    __syncthreads();
#pragma unroll 4
    for (int t = 0; t < KTILE; ++t) {
      float dx = qx - sx[t], dy = qy - sy[t], dz = qz - sz[t];
      float d = fmaf(dx, dx, fmaf(dy, dy, dz * dz));
      int j = base + t;
      if (j == q) d = FLTMAX;               // exclude self
      if (d < worst) {
#pragma unroll
        for (int s = 0; s < KNN_K; ++s)
          if (s == worstSlot) { bd[s] = d; bi[s] = j; }
        worst = bd[0]; worstSlot = 0;
#pragma unroll
        for (int s = 1; s < KNN_K; ++s)
          if (bd[s] > worst) { worst = bd[s]; worstSlot = s; }
      }
    }
  }
#pragma unroll
  for (int s = 0; s < KNN_K; ++s) idxOut[q * KNN_K + s] = bi[s];
}

// ---------------------------------------------------------------------------
// Kernel 2: per-point edge MLP with WMMA. One wave32 per point.
//   ef(16x128) @ W1(128x64) -> ReLU -> (16x64) @ W2(64x64) -> max over 16.
// Block = 128 threads = 4 waves = 4 points. W1^T/W2^T staged in LDS as bf16.
// ---------------------------------------------------------------------------
__global__ void __launch_bounds__(128) edge_mlp_kernel(
    const float* __restrict__ x, const int* __restrict__ nbr,
    const float* __restrict__ W1, const float* __restrict__ b1,
    const float* __restrict__ W2, const float* __restrict__ b2,
    float* __restrict__ out) {
  __shared__ __align__(16) unsigned short w1t[D_HID * 2 * C_IN];   // [64][128] = W1^T
  __shared__ __align__(16) unsigned short w2t[D_HID * D_HID];      // [64][64]  = W2^T
  __shared__ float b1s[D_HID];
  __shared__ float b2s[D_HID];
  __shared__ __align__(16) unsigned short efS[4][KNN_K * 2 * C_IN]; // per-wave [16][128]
  __shared__ __align__(16) unsigned short hS[4][KNN_K * D_HID];     // per-wave [16][64]

  const int tid = threadIdx.x;

  // Cooperative transposed bf16 weight staging.
  for (int t = tid; t < D_HID * 128; t += 128) {
    int d = t >> 7, c = t & 127;
    w1t[t] = f2bf(W1[c * D_HID + d]);
  }
  for (int t = tid; t < D_HID * D_HID; t += 128) {
    int d = t >> 6, c = t & 63;
    w2t[t] = f2bf(W2[c * D_HID + d]);
  }
  if (tid < D_HID) { b1s[tid] = b1[tid]; b2s[tid] = b2[tid]; }
  __syncthreads();

  const int lane   = tid & 31;
  const int laneL  = lane & 15;
  const int laneHi = lane >> 4;
  const int wv     = tid >> 5;
  const int point  = blockIdx.x * 4 + wv;

  unsigned short* ef = efS[wv];
  unsigned short* hs = hS[wv];

  // Per-lane x_i chunk: columns 4*laneL .. 4*laneL+3 (L2-resident gather).
  const float4 xi = *(const float4*)(x + (size_t)point * C_IN + 4 * laneL);
  uint2 xiP;
  xiP.x = ((unsigned)f2bf(xi.x)) | ((unsigned)f2bf(xi.y) << 16);
  xiP.y = ((unsigned)f2bf(xi.z)) | ((unsigned)f2bf(xi.w) << 16);

  int myIdx = nbr[point * KNN_K + laneL];

  // Build ef = [x_i | x_j - x_i] in LDS (bf16, row stride 128).
#pragma unroll
  for (int r = 0; r < KNN_K; ++r) {
    int j = __shfl(myIdx, r);
    if (laneHi) {
      const float4 xj = *(const float4*)(x + (size_t)j * C_IN + 4 * laneL);
      uint2 dp;
      dp.x = ((unsigned)f2bf(xj.x - xi.x)) | ((unsigned)f2bf(xj.y - xi.y) << 16);
      dp.y = ((unsigned)f2bf(xj.z - xi.z)) | ((unsigned)f2bf(xj.w - xi.w) << 16);
      *(uint2*)(ef + r * 128 + C_IN + 4 * laneL) = dp;
    } else {
      *(uint2*)(ef + r * 128 + 4 * laneL) = xiP;
    }
  }
  asm volatile("s_wait_dscnt 0" ::: "memory");   // per-wave LDS fence before operand reads

  // GEMM1: 4 K-steps x 4 N-tiles of v_wmma_f32_16x16x32_bf16.
  v8f acc[4];
#pragma unroll
  for (int nt = 0; nt < 4; ++nt) {
    v8f z = {0.f, 0.f, 0.f, 0.f, 0.f, 0.f, 0.f, 0.f};
    acc[nt] = z;
  }
#pragma unroll
  for (int kt = 0; kt < 4; ++kt) {
    v16bf a = ldsA(ef, laneL, 128, kt, laneHi);
#pragma unroll
    for (int nt = 0; nt < 4; ++nt) {
      v16bf bm = ldsB(w1t, nt * 16 + laneL, 128, kt, laneHi);
      acc[nt] = WMMA_BF16(a, bm, acc[nt]);
    }
  }

  // Bias + ReLU, restage h as bf16 A operand (C-tile: lane=col, vgpr r = rows r/r+8).
  float b1v[4];
#pragma unroll
  for (int nt = 0; nt < 4; ++nt) b1v[nt] = b1s[nt * 16 + laneL];
#pragma unroll
  for (int nt = 0; nt < 4; ++nt) {
#pragma unroll
    for (int r = 0; r < 8; ++r) {
      float v = fmaxf(acc[nt][r] + b1v[nt], 0.0f);
      hs[(r + 8 * laneHi) * D_HID + nt * 16 + laneL] = f2bf(v);
    }
  }
  asm volatile("s_wait_dscnt 0" ::: "memory");

  // GEMM2: 2 K-steps x 4 N-tiles.
  v8f acc2[4];
#pragma unroll
  for (int nt = 0; nt < 4; ++nt) {
    v8f z = {0.f, 0.f, 0.f, 0.f, 0.f, 0.f, 0.f, 0.f};
    acc2[nt] = z;
  }
#pragma unroll
  for (int kt = 0; kt < 2; ++kt) {
    v16bf a = ldsA(hs, laneL, 64, kt, laneHi);
#pragma unroll
    for (int nt = 0; nt < 4; ++nt) {
      v16bf bm = ldsB(w2t, nt * 16 + laneL, 64, kt, laneHi);
      acc2[nt] = WMMA_BF16(a, bm, acc2[nt]);
    }
  }

  // max over the 16 neighbor rows: 8 rows in this lane, 8 in lane^16; bias after max.
#pragma unroll
  for (int nt = 0; nt < 4; ++nt) {
    float m = acc2[nt][0];
#pragma unroll
    for (int r = 1; r < 8; ++r) m = fmaxf(m, acc2[nt][r]);
    m = fmaxf(m, __shfl_xor(m, 16));
    if (laneHi == 0)
      out[(size_t)point * D_HID + nt * 16 + laneL] = m + b2s[nt * 16 + laneL];
  }
}

// ---------------------------------------------------------------------------
extern "C" void kernel_launch(void* const* d_in, const int* in_sizes, int n_in,
                              void* d_out, int out_size, void* d_ws, size_t ws_size,
                              hipStream_t stream) {
  const float* x   = (const float*)d_in[0];
  const float* pos = (const float*)d_in[1];
  const float* W1  = (const float*)d_in[2];
  const float* b1  = (const float*)d_in[3];
  const float* W2  = (const float*)d_in[4];
  const float* b2  = (const float*)d_in[5];
  float* out = (float*)d_out;

  const int N = in_sizes[1] / 3;     // 16384
  int* idxBuf = (int*)d_ws;          // N*16 ints = 1 MB scratch

  knn_kernel<<<N / 128, 128, 0, stream>>>(pos, idxBuf, N);
  edge_mlp_kernel<<<N / 4, 128, 0, stream>>>(x, idxBuf, W1, b1, W2, b2, out);
}
